// GCNLabelEncoderInter_35158602285586
// MI455X (gfx1250) — matrix operand order repaired and checked by reading, not compiled, not run on hardware
//
#include <hip/hip_runtime.h>

#define NN 50000
#define NE 800000
#define DIN 300
#define DHID 400
#define DOUT 512
#define KP1 320   // 300 padded to 32-multiple
#define NP1 448   // 400 padded to 64-multiple
#define KP2 416   // 400 padded to 32-multiple
#define NP2 512

typedef __bf16 v16bf __attribute__((ext_vector_type(16)));
typedef __bf16 v8bf  __attribute__((ext_vector_type(8)));
typedef float  v8f   __attribute__((ext_vector_type(8)));

#ifndef __has_builtin
#define __has_builtin(x) 0
#endif
#if __has_builtin(__builtin_amdgcn_global_load_async_to_lds_b64) && \
    __has_builtin(__builtin_amdgcn_s_wait_asynccnt)
#define USE_ASYNC_LDS 1
typedef int v2i __attribute__((ext_vector_type(2)));
typedef __attribute__((address_space(1))) v2i* as1_v2i_p;  // global (prints as __device__)
typedef __attribute__((address_space(3))) v2i* as3_v2i_p;  // LDS    (prints as __shared__)
#else
#define USE_ASYNC_LDS 0
#endif

__device__ __forceinline__ unsigned short f2bf(float f) {
  unsigned int u = __float_as_uint(f);
  u += 0x7FFFu + ((u >> 16) & 1u);   // round-to-nearest-even
  return (unsigned short)(u >> 16);
}

// ---------------- utility kernels ----------------

__global__ void zero4_kernel(float4* __restrict__ p, long n4) {
  long i = (long)blockIdx.x * blockDim.x + threadIdx.x;
  long stride = (long)gridDim.x * blockDim.x;
  float4 z = make_float4(0.f, 0.f, 0.f, 0.f);
  for (; i < n4; i += stride) p[i] = z;
}

__global__ void count_deg_kernel(const int* __restrict__ src, const int* __restrict__ dst,
                                 float* __restrict__ odeg, float* __restrict__ ideg, int nE) {
  int i = blockIdx.x * blockDim.x + threadIdx.x;
  if (i < nE) {
    atomicAdd(&odeg[src[i]], 1.0f);
    atomicAdd(&ideg[dst[i]], 1.0f);
  }
}

__global__ void rsqrt_deg_kernel(float* __restrict__ d, int n) {
  int i = blockIdx.x * blockDim.x + threadIdx.x;
  if (i < n) {
    float v = d[i];
    v = v < 1.0f ? 1.0f : v;     // DGL clamp
    d[i] = rsqrtf(v);
  }
}

// fp32 W[K,N] row-major -> bf16 TRANSPOSED Wt[Np,Kp] (K contiguous), zero-padded
__global__ void cvt_wt_kernel(const float* __restrict__ W, int K, int N,
                              unsigned short* __restrict__ Wt, int Kp, int Np) {
  int i = blockIdx.x * blockDim.x + threadIdx.x;
  int total = Np * Kp;
  if (i < total) {
    int n = i / Kp, k = i % Kp;
    float v = (k < K && n < N) ? W[k * N + n] : 0.0f;
    Wt[i] = f2bf(v);
  }
}

// ---------------- SpMM: agg[dst] += rout[src] * x[src]  (wave per edge) ----------------
// F4 = features/4 (300->75, 400->100); gathers are float4 (global_load_b128).

__global__ void spmm_kernel(const float* __restrict__ x, int F4,
                            const int* __restrict__ src, const int* __restrict__ dst,
                            const float* __restrict__ rout,
                            float* __restrict__ agg, int strideA, int nE) {
  int gid = blockIdx.x * blockDim.x + threadIdx.x;
  int wave = gid >> 5;
  int lane = threadIdx.x & 31;
  int nw = (gridDim.x * blockDim.x) >> 5;
  for (int e = wave; e < nE; e += nw) {
    int s = src[e];
    int d = dst[e];
    float sc = rout[s];
    const float4* xr = (const float4*)(x + (long)s * (F4 * 4));
    float* ar = agg + (long)d * strideA;
    for (int c = lane; c < F4; c += 32) {
      float4 v = xr[c];
      int b = c * 4;
      atomicAdd(ar + b + 0, sc * v.x);   // f32 atomics resolved in L2
      atomicAdd(ar + b + 1, sc * v.y);
      atomicAdd(ar + b + 2, sc * v.z);
      atomicAdd(ar + b + 3, sc * v.w);
    }
  }
}

// ---------------- dual-branch WMMA GEMM ----------------
// out = f(rowscale1(A1) @ W1 + b1) + f(rowscale2(A2) @ W2 + b2)
// A: f32 [M, Kp] (K zero-padded).  Wt: bf16 [Np, Kp] transposed (K contiguous).
// 256 threads = 8 waves; 64x64 tile; K step 32; A register-prefetch pipeline;
// B tile staged via async global->LDS when available.

__global__ __launch_bounds__(256, 2)
void gemm_dual_kernel(const float* __restrict__ A1, const float* __restrict__ rs1,
                      const unsigned short* __restrict__ W1t, const float* __restrict__ b1,
                      const float* __restrict__ A2, const float* __restrict__ rs2,
                      const unsigned short* __restrict__ W2t, const float* __restrict__ b2,
                      float* __restrict__ out, int M, int N, int ldOut,
                      int Kp, int leaky) {
  __shared__ __align__(16) unsigned short Abuf[64 * 32];  // [m][k] bf16
  __shared__ __align__(16) unsigned short Bbuf[64 * 32];  // [n][k] bf16

  const int m0 = blockIdx.x * 64;
  const int n0 = blockIdx.y * 64;
  const int t = threadIdx.x;
  const int lane = t & 31;
  const int wave = t >> 5;
  const int wm = wave & 3;        // M subtile 0..3
  const int wn = wave >> 2;       // 0..1
  const int nl = lane & 15;
  const int khalf = lane >> 4;

  // fixed per-thread staging coordinates: 64 rows x 32 cols tile, 8 x float4/row
  const int c4 = (t & 7) << 2;
  int row[2], am[2];
  float asc0[2], asc1[2];
  #pragma unroll
  for (int j = 0; j < 2; ++j) {
    row[j] = (t >> 3) + 32 * j;
    int m = m0 + row[j];
    am[j] = (m >= M) ? (M - 1) : m;   // clamp; those outputs never stored
    asc0[j] = rs1[am[j]];
    asc1[j] = rs2[am[j]];
  }

  const int nk = Kp >> 5;
  const int total = nk * 2;

  v8f acc00 = {}, acc01 = {}, acc10 = {}, acc11 = {};
  float4 pA[2];

  auto prefetchA = [&](int s) {
    int g = (s >= nk) ? 1 : 0;
    int k0 = (s - g * nk) << 5;
    const float* A = g ? A2 : A1;
    #pragma unroll
    for (int j = 0; j < 2; ++j)
      pA[j] = *(const float4*)(A + (long)am[j] * Kp + k0 + c4);
  };

  prefetchA(0);

  for (int s = 0; s < total; ++s) {
    const int g = (s >= nk) ? 1 : 0;
    const int k0 = (s - g * nk) << 5;
    const unsigned short* Wt = g ? W2t : W1t;

    // ---- stage tiles (LDS is free here: post-consume barrier of prev iter) ----
#if USE_ASYNC_LDS
    // B tile: pure byte-move global->LDS, ASYNCcnt-tracked, overlaps A staging
    #pragma unroll
    for (int j = 0; j < 2; ++j) {
      const unsigned short* srcp = Wt + (long)(n0 + row[j]) * Kp + k0 + c4;
      unsigned short* dstp = &Bbuf[row[j] * 32 + c4];
      __builtin_amdgcn_global_load_async_to_lds_b64(
          (as1_v2i_p)(unsigned long long)srcp,
          (as3_v2i_p)(unsigned int)(unsigned long long)dstp,
          0, 0);
    }
#endif
    // A tile: f32 -> bf16 with rsqrt(in_deg) row scale, from prefetch registers
    #pragma unroll
    for (int j = 0; j < 2; ++j) {
      float sc = g ? asc1[j] : asc0[j];
      float4 v = pA[j];
      ushort4 p;
      p.x = f2bf(v.x * sc); p.y = f2bf(v.y * sc);
      p.z = f2bf(v.z * sc); p.w = f2bf(v.w * sc);
      *(ushort4*)(&Abuf[row[j] * 32 + c4]) = p;
    }
#if !USE_ASYNC_LDS
    #pragma unroll
    for (int j = 0; j < 2; ++j) {
      ushort4 wv = *(const ushort4*)(Wt + (long)(n0 + row[j]) * Kp + k0 + c4);
      *(ushort4*)(&Bbuf[row[j] * 32 + c4]) = wv;
    }
#endif
    // prefetch next A tile: global loads overlap with this step's WMMAs
    if (s + 1 < total) prefetchA(s + 1);

#if USE_ASYNC_LDS
    __builtin_amdgcn_s_wait_asynccnt(0);
#endif
    __syncthreads();

    // ---- fragments + WMMA ----
    // A frag: lane M = wm*16+nl; K runs [8*khalf..+7], [16+8*khalf..+7]
    int mrow = wm * 16 + nl;
    v8bf a0 = *(const v8bf*)(&Abuf[mrow * 32 + khalf * 8]);
    v8bf a1 = *(const v8bf*)(&Abuf[mrow * 32 + 16 + khalf * 8]);
    v16bf a = __builtin_shufflevector(a0, a1,
                0, 1, 2, 3, 4, 5, 6, 7, 8, 9, 10, 11, 12, 13, 14, 15);
    // B frags: lane N = sub*16+nl; K runs [8*khalf..+7], [16+8*khalf..+7]
    int kb = khalf * 8;
    int nc0 = (wn * 2 + 0) * 16 + nl;
    int nc1 = (wn * 2 + 1) * 16 + nl;
    v8bf q0 = *(const v8bf*)(&Bbuf[nc0 * 32 + kb]);
    v8bf q1 = *(const v8bf*)(&Bbuf[nc0 * 32 + kb + 16]);
    v16bf bf0 = __builtin_shufflevector(q0, q1,
                  0, 1, 2, 3, 4, 5, 6, 7, 8, 9, 10, 11, 12, 13, 14, 15);
    v8bf q2 = *(const v8bf*)(&Bbuf[nc1 * 32 + kb]);
    v8bf q3 = *(const v8bf*)(&Bbuf[nc1 * 32 + kb + 16]);
    v16bf bf1 = __builtin_shufflevector(q2, q3,
                  0, 1, 2, 3, 4, 5, 6, 7, 8, 9, 10, 11, 12, 13, 14, 15);

    if (g == 0) {
      acc00 = __builtin_amdgcn_wmma_f32_16x16x32_bf16(false, a, false, bf0, (short)0, acc00, false, false);
      acc01 = __builtin_amdgcn_wmma_f32_16x16x32_bf16(false, a, false, bf1, (short)0, acc01, false, false);
    } else {
      acc10 = __builtin_amdgcn_wmma_f32_16x16x32_bf16(false, a, false, bf0, (short)0, acc10, false, false);
      acc11 = __builtin_amdgcn_wmma_f32_16x16x32_bf16(false, a, false, bf1, (short)0, acc11, false, false);
    }
    __syncthreads();   // LDS reads done before next iteration overwrites
  }

  // ---- epilogue: bias + (leaky) + branch sum; D layout: M = j + 8*khalf, N = nl ----
  #pragma unroll
  for (int nsub = 0; nsub < 2; ++nsub) {
    int n_g = n0 + (wn * 2 + nsub) * 16 + nl;
    if (n_g >= N) continue;
    float bia = b1[n_g];
    float bib = b2[n_g];
    v8f va = (nsub == 0) ? acc00 : acc01;
    v8f vb = (nsub == 0) ? acc10 : acc11;
    #pragma unroll
    for (int j = 0; j < 8; ++j) {
      int m_g = m0 + wm * 16 + j + 8 * khalf;
      if (m_g >= M) continue;
      float v1 = va[j] + bia;
      float v2 = vb[j] + bib;
      float r;
      if (leaky)
        r = (v1 > 0.f ? v1 : 0.2f * v1) + (v2 > 0.f ? v2 : 0.2f * v2);
      else
        r = v1 + v2;
      out[(long)m_g * ldOut + n_g] = r;
    }
  }
}

// ---------------- host-side orchestration ----------------

extern "C" void kernel_launch(void* const* d_in, const int* in_sizes, int n_in,
                              void* d_out, int out_size, void* d_ws, size_t ws_size,
                              hipStream_t stream) {
  (void)in_sizes; (void)n_in; (void)out_size; (void)ws_size;

  const float* init_emb = (const float*)d_in[0];
  const float* Wa1 = (const float*)d_in[1];
  const float* ba1 = (const float*)d_in[2];
  const float* Wa2 = (const float*)d_in[3];
  const float* ba2 = (const float*)d_in[4];
  const float* Wb1 = (const float*)d_in[5];
  const float* bb1 = (const float*)d_in[6];
  const float* Wb2 = (const float*)d_in[7];
  const float* bb2 = (const float*)d_in[8];
  const int* src1 = (const int*)d_in[9];
  const int* dst1 = (const int*)d_in[10];
  const int* src2 = (const int*)d_in[11];
  const int* dst2 = (const int*)d_in[12];
  float* out = (float*)d_out;

  // workspace carve-up (256B aligned)
  size_t off = 0;
  char* base = (char*)d_ws;
  auto carve = [&](size_t bytes) -> char* {
    char* p = base + off;
    off += (bytes + 255) & ~(size_t)255;
    return p;
  };
  float* rdeg = (float*)carve((size_t)4 * NN * sizeof(float)); // rout1,rin1,rout2,rin2
  unsigned short* wA1t = (unsigned short*)carve((size_t)NP1 * KP1 * 2);
  unsigned short* wB1t = (unsigned short*)carve((size_t)NP1 * KP1 * 2);
  unsigned short* wA2t = (unsigned short*)carve((size_t)NP2 * KP2 * 2);
  unsigned short* wB2t = (unsigned short*)carve((size_t)NP2 * KP2 * 2);
  float* xbuf = (float*)carve((size_t)NN * DHID * sizeof(float));
  float* aggA = (float*)carve((size_t)NN * KP2 * sizeof(float));
  float* aggB = (float*)carve((size_t)NN * KP2 * sizeof(float));  // contiguous after aggA

  float* rout1 = rdeg;
  float* rin1  = rdeg + NN;
  float* rout2 = rdeg + 2 * NN;
  float* rin2  = rdeg + 3 * NN;

  // 1) degrees
  zero4_kernel<<<196, 256, 0, stream>>>((float4*)rdeg, (long)(4 * NN) / 4);
  count_deg_kernel<<<(NE + 255) / 256, 256, 0, stream>>>(src1, dst1, rout1, rin1, NE);
  count_deg_kernel<<<(NE + 255) / 256, 256, 0, stream>>>(src2, dst2, rout2, rin2, NE);
  rsqrt_deg_kernel<<<(4 * NN + 255) / 256, 256, 0, stream>>>(rdeg, 4 * NN);

  // 2) weights -> padded, transposed bf16 (K contiguous)
  cvt_wt_kernel<<<(NP1 * KP1 + 255) / 256, 256, 0, stream>>>(Wa1, DIN, DHID, wA1t, KP1, NP1);
  cvt_wt_kernel<<<(NP1 * KP1 + 255) / 256, 256, 0, stream>>>(Wb1, DIN, DHID, wB1t, KP1, NP1);
  cvt_wt_kernel<<<(NP2 * KP2 + 255) / 256, 256, 0, stream>>>(Wa2, DHID, DOUT, wA2t, KP2, NP2);
  cvt_wt_kernel<<<(NP2 * KP2 + 255) / 256, 256, 0, stream>>>(Wb2, DHID, DOUT, wB2t, KP2, NP2);

  const long aggPairF4 = (long)2 * NN * KP2 / 4;  // aggA+aggB contiguous, zero both

  // 3) hop 1: SpMM (src scale fused) + dual GEMM (dst scale, bias, leaky, sum)
  zero4_kernel<<<4096, 256, 0, stream>>>((float4*)aggA, aggPairF4);
  spmm_kernel<<<2048, 256, 0, stream>>>(init_emb, DIN / 4, src1, dst1, rout1, aggA, KP1, NE);
  spmm_kernel<<<2048, 256, 0, stream>>>(init_emb, DIN / 4, src2, dst2, rout2, aggB, KP1, NE);
  {
    dim3 grid((NN + 63) / 64, (DHID + 63) / 64);
    gemm_dual_kernel<<<grid, 256, 0, stream>>>(aggA, rin1, wA1t, ba1,
                                               aggB, rin2, wB1t, bb1,
                                               xbuf, NN, DHID, DHID, KP1, /*leaky=*/1);
  }

  // 4) hop 2
  zero4_kernel<<<4096, 256, 0, stream>>>((float4*)aggA, aggPairF4);
  spmm_kernel<<<2048, 256, 0, stream>>>(xbuf, DHID / 4, src1, dst1, rout1, aggA, KP2, NE);
  spmm_kernel<<<2048, 256, 0, stream>>>(xbuf, DHID / 4, src2, dst2, rout2, aggB, KP2, NE);
  {
    dim3 grid((NN + 63) / 64, (DOUT + 63) / 64);
    gemm_dual_kernel<<<grid, 256, 0, stream>>>(aggA, rin1, wA2t, ba2,
                                               aggB, rin2, wB2t, bb2,
                                               out, NN, DOUT, DOUT, KP2, /*leaky=*/0);
  }
}
